// NEUMF_Feature_40699110097059
// MI455X (gfx1250) — matrix operand order repaired and checked by reading, not compiled, not run on hardware
//
#include <hip/hip_runtime.h>
#include <hip/hip_bf16.h>

typedef _Float16 half8  __attribute__((ext_vector_type(8)));
typedef _Float16 half16 __attribute__((ext_vector_type(16)));
typedef float    float8 __attribute__((ext_vector_type(8)));

#define NROWS 2048
#define INDIM 512
#define DDIM  64

// ---------------- Kernel 0: fold W2/W3 into u,v (64 each) ----------------
// u[d] = sum_e W2[d,e]   * W3[e]     (alpha weight)
// v[d] = sum_e W2[64+d,e]* W3[e]     (beta weight)   -- both use w3a = W3[:64]
__global__ void uv_kernel(const float* __restrict__ W2, const float* __restrict__ W3,
                          float* __restrict__ u, float* __restrict__ v) {
  int d = threadIdx.x;  // 64 threads
  float su = 0.f, sv = 0.f;
  for (int e = 0; e < DDIM; ++e) {
    float w3a = W3[e];
    su += W2[d * DDIM + e] * w3a;
    sv += W2[(DDIM + d) * DDIM + e] * w3a;
  }
  u[d] = su;
  v[d] = sv;
}

// ---------------- Kernel 1: z = x@W (fp32), emit f16 operands + alpha/beta ----
// Block = 256 threads = 4 rows x 64 cols. x rows staged in LDS (broadcast reads).
__global__ void __launch_bounds__(256)
z_kernel(const float* __restrict__ x, const float* __restrict__ W,
         const float* __restrict__ W3,
         const float* __restrict__ u, const float* __restrict__ v,
         _Float16* __restrict__ zh, _Float16* __restrict__ zwh,
         float* __restrict__ alpha, float* __restrict__ beta) {
  __shared__ float xs[4 * INDIM];
  __shared__ float red[256];
  int t = threadIdx.x;
  int row0 = blockIdx.x * 4;

  for (int idx = t; idx < 4 * INDIM; idx += 256)
    xs[idx] = x[(size_t)row0 * INDIM + idx];
  __syncthreads();

  int rl = t >> 6;      // local row 0..3
  int d  = t & 63;      // feature 0..63
  int row = row0 + rl;

  const float* xr = &xs[rl * INDIM];
  float zf = 0.f;
#pragma unroll 8
  for (int k = 0; k < INDIM; ++k)
    zf = fmaf(xr[k], W[k * DDIM + d], zf);

  zh[(size_t)row * DDIM + d]  = (_Float16)zf;
  zwh[(size_t)row * DDIM + d] = (_Float16)(zf * W3[DDIM + d]);   // w3b = W3[64:]

  float r = zf > 0.f ? zf : 0.f;  // relu(z)
  red[t] = r * u[d];
  __syncthreads();
  if (d == 0) { float s = 0.f; for (int e = 0; e < 64; ++e) s += red[t + e]; alpha[row] = s; }
  __syncthreads();
  red[t] = r * v[d];
  __syncthreads();
  if (d == 0) { float s = 0.f; for (int e = 0; e < 64; ++e) s += red[t + e]; beta[row] = s; }
}

// ---------------- Kernel 2: out = sig(sig(alpha_i + beta_j + z@zw^T)) ----------
// Fast sigmoid: v_mul + v_exp_f32 + v_add + v_rcp_f32 (no IEEE divide sequence).
// v_rcp_f32 is ~1ulp; double-sigmoid compresses any input error by >=16x.
__device__ __forceinline__ float dsig(float xx) {
  return __builtin_amdgcn_rcpf(1.0f + __expf(-xx));
}

// ISA 16-bit A layout (16x32 MxK): lane<16 holds row (lane&15) K{kb..kb+7, kb+16..kb+23},
// lane>=16 holds K{kb+8..kb+15, kb+24..kb+31}. Row-major source -> two 16B loads.
__device__ __forceinline__ half16 load_a_chunk(const _Float16* __restrict__ rowp,
                                               int sel, int kb) {
  half8 lo = *(const half8*)(rowp + kb + sel * 8);
  half8 hi = *(const half8*)(rowp + kb + 16 + sel * 8);
  half16 r;
#pragma unroll
  for (int i = 0; i < 8; ++i) { r[i] = lo[i]; r[i + 8] = hi[i]; }
  return r;
}

__global__ void __launch_bounds__(256)
out_kernel(const _Float16* __restrict__ zh, const _Float16* __restrict__ zwh,
           const float* __restrict__ alpha, const float* __restrict__ beta,
           float* __restrict__ out) {
  int lane = threadIdx.x & 31;
  int wave = threadIdx.x >> 5;
  int strip = blockIdx.x * 8 + wave;   // 4096 strips: 128 i-tiles x 32 j-strips
  int i_tile  = strip >> 5;
  int j_strip = strip & 31;
  int i0 = i_tile * 16;
  int j0 = j_strip * 64;
  int sel = lane >> 4;
  int lr  = lane & 15;

  // A: rows i0..i0+15 of zh, K = 0..63 (two 16x32 chunks)
  const _Float16* arow = zh + (size_t)(i0 + lr) * DDIM;
  half16 a0 = load_a_chunk(arow, sel, 0);
  half16 a1 = load_a_chunk(arow, sel, 32);

  float avals[8];
#pragma unroll
  for (int r = 0; r < 8; ++r) avals[r] = alpha[i0 + 8 * sel + r];

#pragma unroll
  for (int jt = 0; jt < 4; ++jt) {
    // B (32x16 KxN): lane n holds K{kb+8*sel..+7, kb+16+8*sel..+7} of zwh[j0+n][...]
    // -> same K order as A (two 16B loads per chunk).
    const _Float16* brow = zwh + (size_t)(j0 + jt * 16 + lr) * DDIM;
    half16 b0 = load_a_chunk(brow, sel, 0);
    half16 b1 = load_a_chunk(brow, sel, 32);

    float8 c = {0.f, 0.f, 0.f, 0.f, 0.f, 0.f, 0.f, 0.f};
    c = __builtin_amdgcn_wmma_f32_16x16x32_f16(false, a0, false, b0, (short)0, c, false, false);
    c = __builtin_amdgcn_wmma_f32_16x16x32_f16(false, a1, false, b1, (short)0, c, false, false);

    int col = j0 + jt * 16 + lr;
    float bval = beta[col];
#pragma unroll
    for (int r = 0; r < 8; ++r) {
      int rowi = i0 + 8 * sel + r;   // C/D layout: VGPR r -> M = r + 8*sel, N = lane&15
      float vv = c[r] + avals[r] + bval;
      out[(size_t)rowi * NROWS + col] = dsig(dsig(vv));
    }
  }
}

// ---------------- Launch ----------------
extern "C" void kernel_launch(void* const* d_in, const int* in_sizes, int n_in,
                              void* d_out, int out_size, void* d_ws, size_t ws_size,
                              hipStream_t stream) {
  const float* x  = (const float*)d_in[0];   // (2048, 512)
  const float* W  = (const float*)d_in[1];   // (512, 64)
  const float* W2 = (const float*)d_in[2];   // (128, 64)
  const float* W3 = (const float*)d_in[3];   // (128, 1)
  float* out = (float*)d_out;                // (2048, 2048) f32

  char* ws = (char*)d_ws;
  _Float16* zh    = (_Float16*)(ws);               // 2048*64*2 = 256KB
  _Float16* zwh   = (_Float16*)(ws + (256 << 10)); // 256KB
  float*    alpha = (float*)(ws + (512 << 10));    // 8KB
  float*    beta  = (float*)(ws + (520 << 10));    // 8KB
  float*    u     = (float*)(ws + (528 << 10));    // 256B
  float*    v     = (float*)(ws + (529 << 10));    // 256B

  uv_kernel<<<1, 64, 0, stream>>>(W2, W3, u, v);
  z_kernel<<<NROWS / 4, 256, 0, stream>>>(x, W, W3, u, v, zh, zwh, alpha, beta);
  out_kernel<<<(128 * 32) / 8, 256, 0, stream>>>(zh, zwh, alpha, beta, out);
}
